// CausalAttention_85529978732635
// MI455X (gfx1250) — matrix operand order repaired and checked by reading, not compiled, not run on hardware
//
#include <hip/hip_runtime.h>
#include <hip/hip_bf16.h>
#include <cstdint>

// ---------------------------------------------------------------------------
// CDNA5 (gfx1250) causal attention: bf16 WMMA GEMMs + fp32 softmax.
// Pipeline: convert -> Q/K/V projections -> QK^T (causal-skip) -> softmax -> PV
// GEMM inner loop: 8 K-steps fully unrolled with compile-time load offsets
// (IOFFSET immediates); loop-carried fragment pointers advance once per outer
// iteration. K must be a multiple of 256 (here 1024 or 2048).
// ---------------------------------------------------------------------------

typedef __attribute__((ext_vector_type(16))) __bf16 v16bf;
typedef __attribute__((ext_vector_type(8)))  float  v8f;

static_assert(sizeof(v16bf) == 32, "v16bf must be 32 bytes");

__device__ __forceinline__ unsigned short f32_to_bf16(float f) {
    unsigned int u = __builtin_bit_cast(unsigned int, f);
    // round-to-nearest-even
    unsigned int r = u + 0x7FFFu + ((u >> 16) & 1u);
    return (unsigned short)(r >> 16);
}

union FragBF16 {
    uint4 q[2];
    v16bf v;
};

// A fragment: 16-bit A 16x32 (ISA 7.12.2): two 16B chunks, 16 elements apart.
__device__ __forceinline__ void load_frag_a(FragBF16& f, const unsigned short* p) {
    f.q[0] = *reinterpret_cast<const uint4*>(p);
    f.q[1] = *reinterpret_cast<const uint4*>(p + 16);
}
// B fragment: 16-bit B 32x16: two 16B chunks of one column, 16 elements apart.
__device__ __forceinline__ void load_frag_b(FragBF16& f, const unsigned short* p) {
    f.q[0] = *reinterpret_cast<const uint4*>(p);
    f.q[1] = *reinterpret_cast<const uint4*>(p + 16);
}

// ---------------------------------------------------------------------------
// Generic bf16 x bf16 -> f32-acc WMMA GEMM.
//   C[m,n] = sum_k A[m,k] * Bcol[n,k]
// A:  row-major [M x K], lda elements.
// Bm: B operand stored so column n of the K x N matrix is contiguous:
//     Bm + n*ldb + k  ("B^T row-major").
// MODE 0: store bf16 row-major (ldo)
// MODE 1: store bf16 transposed per folded batch: out[b][n][s], s = m % Sper
// MODE 2: store f32 row-major (ldo)
// CAUSAL: skip tiles entirely above the diagonal (scores GEMM).
// grid: (N/64, M/128, nbatch), block: 128 threads (4 waves, each a 32x64 tile)
// Requires: K % 256 == 0.
// ---------------------------------------------------------------------------
template <int MODE, bool CAUSAL>
__global__ __launch_bounds__(128)
void gemm_bf16_wmma(const unsigned short* __restrict__ A, long long strideA, int lda,
                    const unsigned short* __restrict__ Bm, long long strideB, int ldb,
                    void* __restrict__ Out, long long strideO, int ldo,
                    int M, int N, int K, int Sper) {
    const int lane = threadIdx.x & 31;
    const int wave = threadIdx.x >> 5;
    const int m0 = (int)blockIdx.y * 128 + wave * 32;  // wave handles rows m0..m0+31
    const int n0 = (int)blockIdx.x * 64;

    if (CAUSAL && n0 > (int)blockIdx.y * 128 + 127) return;  // fully masked tile

    A  += (long long)blockIdx.z * strideA;
    Bm += (long long)blockIdx.z * strideB;

    v8f acc[2][4] = {};

    // A fragment lane mapping: lanes 0-15 row=lane, K={0..7,16..23};
    //                          lanes 16-31 row=lane-16, K={8..15,24..31}
    const int arow = m0 + (lane & 15);
    const int akb  = (lane & 16) ? 8 : 0;
    // B fragment lane mapping: lanes 0-15 col=lane K={0..7,16..23}; lanes 16-31 K={8..15,24..31}
    const int bcol = lane & 15;
    const int bkb  = (lane & 16) ? 8 : 0;

    // loop-carried fragment pointers: computed once, advanced 256 elems/outer iter
    const unsigned short* pa0 = A + (long long)arow * lda + akb;
    const unsigned short* pa1 = pa0 + (long long)16 * lda;
    const unsigned short* pb0 = Bm + (long long)(n0 + 0 * 16 + bcol) * ldb + bkb;
    const unsigned short* pb1 = Bm + (long long)(n0 + 1 * 16 + bcol) * ldb + bkb;
    const unsigned short* pb2 = Bm + (long long)(n0 + 2 * 16 + bcol) * ldb + bkb;
    const unsigned short* pb3 = Bm + (long long)(n0 + 3 * 16 + bcol) * ldb + bkb;

    for (int kk = 0; kk < K; kk += 256) {
        __builtin_prefetch(pa0 + 256, 0, 0);  // global_prefetch, next A block
#pragma unroll
        for (int u = 0; u < 8; ++u) {
            const int ko = u * 32;  // compile-time -> folds into load IOFFSET
            FragBF16 a0, a1, b0, b1, b2, b3;
            load_frag_a(a0, pa0 + ko);
            load_frag_a(a1, pa1 + ko);
            load_frag_b(b0, pb0 + ko);
            load_frag_b(b1, pb1 + ko);
            load_frag_b(b2, pb2 + ko);
            load_frag_b(b3, pb3 + ko);

            acc[0][0] = __builtin_amdgcn_wmma_f32_16x16x32_bf16(
                false, a0.v, false, b0.v, (short)0, acc[0][0], false, false);
            acc[1][0] = __builtin_amdgcn_wmma_f32_16x16x32_bf16(
                false, a1.v, false, b0.v, (short)0, acc[1][0], false, false);
            acc[0][1] = __builtin_amdgcn_wmma_f32_16x16x32_bf16(
                false, a0.v, false, b1.v, (short)0, acc[0][1], false, false);
            acc[1][1] = __builtin_amdgcn_wmma_f32_16x16x32_bf16(
                false, a1.v, false, b1.v, (short)0, acc[1][1], false, false);
            acc[0][2] = __builtin_amdgcn_wmma_f32_16x16x32_bf16(
                false, a0.v, false, b2.v, (short)0, acc[0][2], false, false);
            acc[1][2] = __builtin_amdgcn_wmma_f32_16x16x32_bf16(
                false, a1.v, false, b2.v, (short)0, acc[1][2], false, false);
            acc[0][3] = __builtin_amdgcn_wmma_f32_16x16x32_bf16(
                false, a0.v, false, b3.v, (short)0, acc[0][3], false, false);
            acc[1][3] = __builtin_amdgcn_wmma_f32_16x16x32_bf16(
                false, a1.v, false, b3.v, (short)0, acc[1][3], false, false);
        }
        pa0 += 256; pa1 += 256;
        pb0 += 256; pb1 += 256; pb2 += 256; pb3 += 256;
    }

    // D layout: lanes 0-15 rows 0..7 (VGPR i = row i), lanes 16-31 rows 8..15.
    const int srow = (lane & 16) ? 8 : 0;
    const int scol = lane & 15;
#pragma unroll
    for (int mi = 0; mi < 2; ++mi) {
#pragma unroll
        for (int j = 0; j < 4; ++j) {
#pragma unroll
            for (int i = 0; i < 8; ++i) {
                const int r = m0 + mi * 16 + srow + i;
                const int c = n0 + j * 16 + scol;
                const float val = acc[mi][j][i];
                if (MODE == 0) {
                    unsigned short* o =
                        (unsigned short*)Out + (long long)blockIdx.z * strideO;
                    o[(long long)r * ldo + c] = f32_to_bf16(val);
                } else if (MODE == 1) {
                    unsigned short* o = (unsigned short*)Out;
                    const int bg = r / Sper;
                    const int s  = r % Sper;
                    o[(long long)bg * N * Sper + (long long)c * Sper + s] =
                        f32_to_bf16(val);
                } else {
                    float* o = (float*)Out + (long long)blockIdx.z * strideO;
                    o[(long long)r * ldo + c] = val;
                }
            }
        }
    }
}

// ---------------------------------------------------------------------------
// Row-wise causal softmax: P[b,q,k] = softmax_k(scores[b,q,k]*scale | k<=q)
// grid: (S, B), block 256; S must equal 8*256 = 2048.
// ---------------------------------------------------------------------------
__global__ __launch_bounds__(256)
void softmax_causal(const float* __restrict__ scores,
                    unsigned short* __restrict__ P, int S, float scale) {
    const int q = blockIdx.x;
    const int b = blockIdx.y;
    const long long roff = ((long long)b * S + q) * (long long)S;
    const float* row = scores + roff;
    unsigned short* prow = P + roff;

    __shared__ float red[256];

    float vals[8];
    float mx = -3.4e38f;
#pragma unroll
    for (int i = 0; i < 8; ++i) {
        const int k = (int)threadIdx.x + i * 256;
        const float v = (k <= q) ? row[k] * scale : -3.4e38f;
        vals[i] = v;
        mx = fmaxf(mx, v);
    }
    red[threadIdx.x] = mx;
    __syncthreads();
    for (int s2 = 128; s2 > 0; s2 >>= 1) {
        if ((int)threadIdx.x < s2)
            red[threadIdx.x] = fmaxf(red[threadIdx.x], red[threadIdx.x + s2]);
        __syncthreads();
    }
    mx = red[0];
    __syncthreads();

    float sum = 0.f;
#pragma unroll
    for (int i = 0; i < 8; ++i) {
        const int k = (int)threadIdx.x + i * 256;
        const float e = (k <= q) ? __expf(vals[i] - mx) : 0.f;
        vals[i] = e;
        sum += e;
    }
    red[threadIdx.x] = sum;
    __syncthreads();
    for (int s2 = 128; s2 > 0; s2 >>= 1) {
        if ((int)threadIdx.x < s2) red[threadIdx.x] += red[threadIdx.x + s2];
        __syncthreads();
    }
    const float inv = 1.f / red[0];

#pragma unroll
    for (int i = 0; i < 8; ++i) {
        const int k = (int)threadIdx.x + i * 256;
        prow[k] = f32_to_bf16(vals[i] * inv);
    }
}

// ---------------------------------------------------------------------------
// fp32 -> bf16 raw conversion (grid-stride)
// ---------------------------------------------------------------------------
__global__ void convert_f32_to_bf16(const float* __restrict__ src,
                                    unsigned short* __restrict__ dst,
                                    long long n) {
    long long i = (long long)blockIdx.x * blockDim.x + threadIdx.x;
    const long long stride = (long long)gridDim.x * blockDim.x;
    for (; i < n; i += stride) dst[i] = f32_to_bf16(src[i]);
}

// ---------------------------------------------------------------------------
// Launch
// ---------------------------------------------------------------------------
extern "C" void kernel_launch(void* const* d_in, const int* in_sizes, int n_in,
                              void* d_out, int out_size, void* d_ws, size_t ws_size,
                              hipStream_t stream) {
    (void)in_sizes; (void)n_in; (void)out_size; (void)ws_size;

    constexpr int B = 4, S = 2048, D = 1024;
    constexpr long long MS = (long long)B * S;  // folded rows for projections

    const float* x  = (const float*)d_in[0];
    const float* Wq = (const float*)d_in[1];
    const float* Wk = (const float*)d_in[2];
    const float* Wv = (const float*)d_in[3];
    float* out = (float*)d_out;

    // workspace carve-up (256B aligned)
    char* p = (char*)d_ws;
    auto alloc = [&](size_t bytes) -> void* {
        void* r = (void*)p;
        p += (bytes + 255) & ~(size_t)255;
        return r;
    };
    unsigned short* xb  = (unsigned short*)alloc((size_t)MS * D * 2);      // 16 MB
    unsigned short* wqb = (unsigned short*)alloc((size_t)D * D * 2);       //  2 MB
    unsigned short* wkb = (unsigned short*)alloc((size_t)D * D * 2);       //  2 MB
    unsigned short* wvb = (unsigned short*)alloc((size_t)D * D * 2);       //  2 MB
    unsigned short* Qb  = (unsigned short*)alloc((size_t)MS * D * 2);      // 16 MB
    unsigned short* Kb  = (unsigned short*)alloc((size_t)MS * D * 2);      // 16 MB
    unsigned short* Vtb = (unsigned short*)alloc((size_t)MS * D * 2);      // 16 MB [b][e][s]
    float*          scr = (float*)alloc((size_t)B * S * S * 4);            // 64 MB
    unsigned short* Pb  = (unsigned short*)alloc((size_t)B * S * S * 2);   // 32 MB

    // 1) convert inputs to bf16
    convert_f32_to_bf16<<<2048, 256, 0, stream>>>(x, xb, MS * (long long)D);
    convert_f32_to_bf16<<<1024, 256, 0, stream>>>(Wq, wqb, (long long)D * D);
    convert_f32_to_bf16<<<1024, 256, 0, stream>>>(Wk, wkb, (long long)D * D);
    convert_f32_to_bf16<<<1024, 256, 0, stream>>>(Wv, wvb, (long long)D * D);

    // 2) projections: Q/K row-major bf16, V transposed per batch.
    //    y[m,e] = sum_d xb[m,d] * W[e,d]  -> B-operand column e == W row e (contig)
    {
        dim3 grid(D / 64, (int)(MS / 128), 1);
        gemm_bf16_wmma<0, false><<<grid, 128, 0, stream>>>(
            xb, 0, D, wqb, 0, D, Qb, 0, D, (int)MS, D, D, S);
        gemm_bf16_wmma<0, false><<<grid, 128, 0, stream>>>(
            xb, 0, D, wkb, 0, D, Kb, 0, D, (int)MS, D, D, S);
        gemm_bf16_wmma<1, false><<<grid, 128, 0, stream>>>(
            xb, 0, D, wvb, 0, D, Vtb, 0, S, (int)MS, D, D, S);
    }

    // 3) scores[b,q,k] = sum_d Q[b,q,d] K[b,k,d]   (causal tile skipping)
    {
        dim3 grid(S / 64, S / 128, B);
        gemm_bf16_wmma<2, true><<<grid, 128, 0, stream>>>(
            Qb, (long long)S * D, D,
            Kb, (long long)S * D, D,
            scr, (long long)S * S, S,
            S, S, D, S);
    }

    // 4) masked softmax with 1/sqrt(D) scaling -> bf16 probabilities
    softmax_causal<<<dim3(S, B), 256, 0, stream>>>(scr, Pb, S, 0.03125f);

    // 5) out[b,s,e] = sum_k P[b,s,k] * V[b,k,e]; Vt row e contiguous in k
    {
        dim3 grid(D / 64, S / 128, B);
        gemm_bf16_wmma<2, false><<<grid, 128, 0, stream>>>(
            Pb, (long long)S * S, S,
            Vtb, (long long)D * S, S,
            out, (long long)S * D, D,
            S, D, S, S);
    }
}
